// AttentionBlock_19258633355634
// MI455X (gfx1250) — matrix operand order repaired and checked by reading, not compiled, not run on hardware
//
#include <hip/hip_runtime.h>

// ---------------------------------------------------------------------------
// AttentionBlock for MI455X (gfx1250, wave32, WMMA 16x16x32 f16 -> f32)
//   B=32, C=512, H=W=32, N=1024, GROUPS=32 (16 ch/group)
// Pipeline: groupnorm -> QKV gemm -> q^T k gemm -> softmax -> P v gemm
//           -> proj gemm + bias + residual
// GEMM: 128x128x32 tiles, 8 waves/block, double-buffered LDS fed by
// GLOBAL_LOAD_ASYNC_TO_LDS_B128 (ASYNCcnt), WMMA f16->f32 accumulate.
// ---------------------------------------------------------------------------

typedef __attribute__((ext_vector_type(16))) _Float16 v16h;
typedef __attribute__((ext_vector_type(8)))  _Float16 v8h;
typedef __attribute__((ext_vector_type(8)))  float    v8f;

#define LDS_PITCH 40   // 32 K-halfs + 8 pad (80B rows, 16B aligned)
#define BUF_HALFS (128 * LDS_PITCH)
#define BC 512
#define BN 1024
#define Q_SCALE 0.044194173824159216f   // 512^-0.5

// ---- CDNA5 async global->LDS copy (16B per lane, tracked by ASYNCcnt) -----
__device__ __forceinline__ void async_copy_b128(const _Float16* g, _Float16* l) {
  unsigned lds = (unsigned)(unsigned long long)l;  // low 32b = LDS offset
  asm volatile("global_load_async_to_lds_b128 %0, %1, off"
               :: "v"(lds), "v"(g) : "memory");
}
template <int N>
__device__ __forceinline__ void wait_asynccnt() {
  asm volatile("s_wait_asynccnt %0" :: "i"(N) : "memory");
}

// ---- WMMA fragment loads from LDS (per ISA 16-bit VGPR layouts) -----------
__device__ __forceinline__ v16h load_a_frag(const _Float16* tileBase) {
  const int lane = threadIdx.x & 31;
  const int m = lane & 15, hi = lane >> 4;
  const _Float16* row = tileBase + m * LDS_PITCH;
  v8h lo = *(const v8h*)(row + hi * 8);        // k = hi*8 .. hi*8+7
  v8h hv = *(const v8h*)(row + 16 + hi * 8);   // k = 16+hi*8 ..
  v16h r;
#pragma unroll
  for (int i = 0; i < 8; ++i) { r[i] = lo[i]; r[i + 8] = hv[i]; }
  return r;
}

__device__ __forceinline__ v16h load_b_frag(const _Float16* tileBase) {
  const int lane = threadIdx.x & 31;
  const int n = lane & 15, hi = lane >> 4;
  const _Float16* row = tileBase + n * LDS_PITCH;  // k = hi*8 + t, 16 + hi*8 + t
  v8h lo = *(const v8h*)(row + hi * 8);
  v8h hv = *(const v8h*)(row + 16 + hi * 8);
  v16h r;
#pragma unroll
  for (int i = 0; i < 8; ++i) { r[i] = lo[i]; r[i + 8] = hv[i]; }
  return r;
}

// ---- Generic tiled GEMM: D(m,n) = sum_k A[m,k] * Bt[n,k] ------------------
// A row-major (M x K, pitch lda), Bt row-major (N x K, pitch ldb), f16.
// Block tile 128x128, K-step 32, 8 waves (2x4), each wave 64x32 (4x2 frags).
template <class Epi>
__global__ __launch_bounds__(256) void gemm_f16_kernel(
    const _Float16* __restrict__ A, long long aBatch, int lda,
    const _Float16* __restrict__ Bt, long long bBatch, int ldb,
    int K, Epi epi) {
  __shared__ _Float16 sA[2 * BUF_HALFS];
  __shared__ _Float16 sB[2 * BUF_HALFS];
  const int t = threadIdx.x;
  const int lane = t & 31, wave = t >> 5;
  const int wm = wave >> 2, wn = wave & 3;
  const int bz = blockIdx.z;
  const int m0 = blockIdx.y * 128, n0 = blockIdx.x * 128;

  const int srow = t >> 1;            // 0..127 (one half-row per thread)
  const int soff = (t & 1) * 16;      // halfs
  const _Float16* gA = A + (long long)bz * aBatch + (long long)(m0 + srow) * lda + soff;
  const _Float16* gB = Bt + (long long)bz * bBatch + (long long)(n0 + srow) * ldb + soff;
  _Float16* lA = sA + srow * LDS_PITCH + soff;
  _Float16* lB = sB + srow * LDS_PITCH + soff;

  v8f acc[4][2];
#pragma unroll
  for (int i = 0; i < 4; ++i)
#pragma unroll
    for (int j = 0; j < 2; ++j)
#pragma unroll
      for (int r = 0; r < 8; ++r) acc[i][j][r] = 0.0f;

  // prefetch tile 0 into buffer 0 (4 async instructions per wave)
  async_copy_b128(gA, lA);
  async_copy_b128(gA + 8, lA + 8);
  async_copy_b128(gB, lB);
  async_copy_b128(gB + 8, lB + 8);

  int buf = 0;
  for (int k0 = 0; k0 < K; k0 += 32) {
    if (k0 + 32 < K) {  // prefetch next tile into the other buffer
      const int nb = (buf ^ 1) * BUF_HALFS;
      async_copy_b128(gA + k0 + 32, lA + nb);
      async_copy_b128(gA + k0 + 40, lA + nb + 8);
      async_copy_b128(gB + k0 + 32, lB + nb);
      async_copy_b128(gB + k0 + 40, lB + nb + 8);
      wait_asynccnt<4>();   // current tile's 4 loads done; next 4 in flight
    } else {
      wait_asynccnt<0>();
    }
    __syncthreads();        // all waves' current tile resident in LDS

    const _Float16* tA = sA + buf * BUF_HALFS;
    const _Float16* tB = sB + buf * BUF_HALFS;
    v16h af[4], bf[2];
#pragma unroll
    for (int fm = 0; fm < 4; ++fm)
      af[fm] = load_a_frag(tA + (wm * 64 + fm * 16) * LDS_PITCH);
#pragma unroll
    for (int fn = 0; fn < 2; ++fn)
      bf[fn] = load_b_frag(tB + (wn * 32 + fn * 16) * LDS_PITCH);
#pragma unroll
    for (int fm = 0; fm < 4; ++fm)
#pragma unroll
      for (int fn = 0; fn < 2; ++fn)
        acc[fm][fn] = __builtin_amdgcn_wmma_f32_16x16x32_f16(
            false, af[fm], false, bf[fn], (short)0, acc[fm][fn], false, false);
    __syncthreads();        // done reading buf; safe to overwrite next iter
    buf ^= 1;
  }

  const int hi = lane >> 4, nn = lane & 15;
#pragma unroll
  for (int fm = 0; fm < 4; ++fm)
#pragma unroll
    for (int fn = 0; fn < 2; ++fn) {
      const int mb = m0 + wm * 64 + fm * 16 + hi * 8;
      const int nb = n0 + wn * 32 + fn * 16 + nn;
#pragma unroll
      for (int r = 0; r < 8; ++r) epi(bz, mb + r, nb, acc[fm][fn][r]);
    }
}

// ---- Epilogues ------------------------------------------------------------
struct QKVEpi {   // split qkv; q -> transposed+scaled, k -> transposed, v -> (C,N)
  _Float16* qt; _Float16* kt; _Float16* vv; const float* bias;
  __device__ void operator()(int b, int m, int n, float val) const {
    val += bias[m];
    if (m < 512)
      qt[(long long)b * 524288 + (long long)n * 512 + m] = (_Float16)(val * Q_SCALE);
    else if (m < 1024)
      kt[(long long)b * 524288 + (long long)n * 512 + (m - 512)] = (_Float16)val;
    else
      vv[(long long)b * 524288 + (long long)(m - 1024) * 1024 + n] = (_Float16)val;
  }
};
struct ScoreEpi { // raw scores, f32, row pitch 1024
  float* S;
  __device__ void operator()(int b, int m, int n, float val) const {
    S[(long long)b * 1048576 + (long long)m * 1024 + n] = val;
  }
};
struct OEpi {     // attention output transposed (N x C), f16
  _Float16* ot;
  __device__ void operator()(int b, int m, int n, float val) const {
    ot[(long long)b * 524288 + (long long)m * 512 + n] = (_Float16)val;
  }
};
struct ProjEpi {  // proj + bias + residual, f32 out (chunk-local pointers)
  float* out; const float* x; const float* bias;
  __device__ void operator()(int b, int m, int n, float val) const {
    const long long idx = (long long)b * 524288 + (long long)m * 1024 + n;
    out[idx] = val + bias[m] + x[idx];
  }
};

// ---- GroupNorm: per (group, batch) block; writes h transposed (N x C) f16 -
__global__ __launch_bounds__(256) void groupnorm_kernel(
    const float* __restrict__ x, const float* __restrict__ gw,
    const float* __restrict__ gb, _Float16* __restrict__ ht) {
  const int g = blockIdx.x;        // 0..31
  const int b = blockIdx.y;        // chunk-local batch
  const float* xg = x + ((long long)b * BC + g * 16) * BN;  // 16 ch x 1024 px
  const int t = threadIdx.x;
  float s = 0.f, ss = 0.f;
  for (int i = t; i < 16384; i += 256) { float v = xg[i]; s += v; ss += v * v; }
  __shared__ float r1[256], r2[256];
  r1[t] = s; r2[t] = ss;
  __syncthreads();
  for (int w = 128; w > 0; w >>= 1) {
    if (t < w) { r1[t] += r1[t + w]; r2[t] += r2[t + w]; }
    __syncthreads();
  }
  const float mean = r1[0] * (1.0f / 16384.0f);
  const float var  = r2[0] * (1.0f / 16384.0f) - mean * mean;
  const float rstd = rsqrtf(var + 1e-5f);
  float wv[16], bv[16];
#pragma unroll
  for (int c = 0; c < 16; ++c) { wv[c] = gw[g * 16 + c]; bv[c] = gb[g * 16 + c]; }
  for (int p = t; p < BN; p += 256) {
    v8h o0, o1;
#pragma unroll
    for (int c = 0; c < 8; ++c) {
      o0[c] = (_Float16)((xg[c * BN + p] - mean) * rstd * wv[c] + bv[c]);
      o1[c] = (_Float16)((xg[(c + 8) * BN + p] - mean) * rstd * wv[c + 8] + bv[c + 8]);
    }
    _Float16* dst = ht + (long long)b * 524288 + (long long)p * BC + g * 16;
    *(v8h*)dst = o0;
    *(v8h*)(dst + 8) = o1;
  }
}

// ---- Row softmax: reads 1024 f32, overwrites row start with 1024 f16 P ----
__global__ __launch_bounds__(256) void softmax_kernel(float* __restrict__ S) {
  const long long row = (long long)blockIdx.y * 1024 + blockIdx.x;
  float* Sr = S + row * 1024;
  const int t = threadIdx.x;
  float v[4];
#pragma unroll
  for (int i = 0; i < 4; ++i) v[i] = Sr[t + i * 256];
  __shared__ float red[256];
  float mx = fmaxf(fmaxf(v[0], v[1]), fmaxf(v[2], v[3]));
  red[t] = mx;
  __syncthreads();
  for (int w = 128; w > 0; w >>= 1) {
    if (t < w) red[t] = fmaxf(red[t], red[t + w]);
    __syncthreads();
  }
  const float m = red[0];
  __syncthreads();
  float e[4], sum = 0.f;
#pragma unroll
  for (int i = 0; i < 4; ++i) { e[i] = __expf(v[i] - m); sum += e[i]; }
  red[t] = sum;
  __syncthreads();
  for (int w = 128; w > 0; w >>= 1) {
    if (t < w) red[t] += red[t + w];
    __syncthreads();
  }
  const float inv = 1.0f / red[0];
  _Float16* P = (_Float16*)Sr;      // in-place: all reads happened pre-barrier
#pragma unroll
  for (int i = 0; i < 4; ++i) P[t + i * 256] = (_Float16)(e[i] * inv);
}

// ---- f32 -> f16 weight conversion -----------------------------------------
__global__ void cvt_f16_kernel(const float* __restrict__ s,
                               _Float16* __restrict__ d, int n) {
  int i = blockIdx.x * 256 + threadIdx.x;
  if (i < n) d[i] = (_Float16)s[i];
}

// ---------------------------------------------------------------------------
extern "C" void kernel_launch(void* const* d_in, const int* in_sizes, int n_in,
                              void* d_out, int out_size, void* d_ws, size_t ws_size,
                              hipStream_t stream) {
  const float* x      = (const float*)d_in[0];
  const float* gn_w   = (const float*)d_in[1];
  const float* gn_b   = (const float*)d_in[2];
  const float* qkv_w  = (const float*)d_in[3];
  const float* qkv_b  = (const float*)d_in[4];
  const float* proj_w = (const float*)d_in[5];
  const float* proj_b = (const float*)d_in[6];
  float* out = (float*)d_out;

  // batch-chunk size from available workspace (~9 MB/batch + 2 MB fixed)
  const size_t perBatch = 5ull * 1024 * 512 * 2 /*h,qt,kt,v,ot*/ +
                          4ull * 1024 * 1024   /*S*/ + 8 * 256;
  const size_t fixed = (size_t)1536 * 512 * 2 + (size_t)512 * 512 * 2 + 4 * 256;
  int cb = 8;
  while (cb > 1 && fixed + (size_t)cb * perBatch > ws_size) cb >>= 1;

  char* p = (char*)d_ws;
  auto carve = [&](size_t bytes) -> void* {
    void* r = (void*)p;
    p += (bytes + 255) & ~(size_t)255;
    return r;
  };
  _Float16* wQ = (_Float16*)carve((size_t)1536 * 512 * 2);
  _Float16* wP = (_Float16*)carve((size_t)512 * 512 * 2);
  _Float16* hT = (_Float16*)carve((size_t)cb * 1024 * 512 * 2);
  _Float16* qT = (_Float16*)carve((size_t)cb * 1024 * 512 * 2);
  _Float16* kT = (_Float16*)carve((size_t)cb * 1024 * 512 * 2);
  _Float16* vB = (_Float16*)carve((size_t)cb * 1024 * 512 * 2);
  _Float16* oT = (_Float16*)carve((size_t)cb * 1024 * 512 * 2);
  float*    S  = (float*)   carve((size_t)cb * 1024 * 1024 * 4);

  cvt_f16_kernel<<<(1536 * 512 + 255) / 256, 256, 0, stream>>>(qkv_w, wQ, 1536 * 512);
  cvt_f16_kernel<<<(512 * 512 + 255) / 256, 256, 0, stream>>>(proj_w, wP, 512 * 512);

  for (int b0 = 0; b0 < 32; b0 += cb) {
    const float* xc = x + (long long)b0 * BC * BN;
    float* oc = out + (long long)b0 * BC * BN;

    groupnorm_kernel<<<dim3(32, cb), 256, 0, stream>>>(xc, gn_w, gn_b, hT);

    // qkv = W(1536x512) x h(512x1024);  Bt = hT (N x C)
    gemm_f16_kernel<QKVEpi><<<dim3(8, 12, cb), 256, 0, stream>>>(
        wQ, 0, 512, hT, 524288, 512, 512, QKVEpi{qT, kT, vB, qkv_b});

    // S = q^T k  (1024 x 1024, K=512);  A = qT, Bt = kT
    gemm_f16_kernel<ScoreEpi><<<dim3(8, 8, cb), 256, 0, stream>>>(
        qT, 524288, 512, kT, 524288, 512, 512, ScoreEpi{S});

    softmax_kernel<<<dim3(1024, cb), 256, 0, stream>>>(S);

    // O^T = P(1024x1024, f16 pitched over S rows) x v^T;  Bt = v (C x N)
    gemm_f16_kernel<OEpi><<<dim3(4, 8, cb), 256, 0, stream>>>(
        (const _Float16*)S, 2097152, 2048, vB, 524288, 1024, 1024, OEpi{oT});

    // out = proj_w(512x512) x O + bias + x;  Bt = oT (N x C)
    gemm_f16_kernel<ProjEpi><<<dim3(8, 4, cb), 256, 0, stream>>>(
        wP, 0, 512, oT, 524288, 512, 512, ProjEpi{oc, xc, proj_b});
  }
  (void)in_sizes; (void)n_in; (void)out_size;
}
